// TransformerCombinator_33457795236050
// MI455X (gfx1250) — matrix operand order repaired and checked by reading, not compiled, not run on hardware
//
#include <hip/hip_runtime.h>
#include <cstdint>
#include <cstddef>

typedef _Float16 half_t;
typedef __attribute__((ext_vector_type(16))) _Float16 v16h;
typedef __attribute__((ext_vector_type(8)))  float    v8f;

constexpr int D     = 512;
constexpr int B     = 64;
constexpr int NROWS = 131072;
constexpr int CAP   = 2304;                 // 18*128; >= max_n with huge margin
constexpr size_t TCAP = (size_t)B * CAP;

// meta layout (ints)
constexpr int M_COUNTS = 0;    // [64]
constexpr int M_STARTS = 64;   // [64]
constexpr int M_MAXN   = 128;

// workspace byte offsets
constexpr size_t OFF_META = 0;
constexpr size_t OFF_WINH = 1024;
constexpr size_t SZ_WINH  = (size_t)3*D*D*2;
constexpr size_t OFF_WOH  = OFF_WINH + SZ_WINH;
constexpr size_t SZ_WOH   = (size_t)D*D*2;
constexpr size_t OFF_W1H  = OFF_WOH + SZ_WOH;
constexpr size_t SZ_W1H   = (size_t)2*D*D*2;
constexpr size_t OFF_W2H  = OFF_W1H + SZ_W1H;
constexpr size_t SZ_W2H   = (size_t)2*D*D*2;
constexpr size_t OFF_XF   = OFF_W2H + SZ_W2H;          // f32 residual x
constexpr size_t SZ_XF    = TCAP * D * 4;
constexpr size_t OFF_XH   = OFF_XF + SZ_XF;            // f16 copy of x
constexpr size_t SZ_XH    = TCAP * D * 2;
constexpr size_t OFF_QKV  = OFF_XH + SZ_XH;            // f16 qkv
constexpr size_t SZ_QKV   = TCAP * 3*D * 2;
constexpr size_t OFF_AO   = OFF_QKV + SZ_QKV;          // f16 attn out
constexpr size_t SZ_AO    = TCAP * D * 2;
constexpr size_t OFF_H1   = OFF_AO + SZ_AO;            // f16 relu hidden
constexpr size_t SZ_H1    = TCAP * 2*D * 2;
constexpr size_t OFF_ZF   = OFF_H1 + SZ_H1;            // f32 pre-LN buffer
constexpr size_t SZ_ZF    = TCAP * D * 4;

static __device__ __forceinline__ unsigned ldsu(const half_t* p) {
    return *(const unsigned*)p;
}

// Low 32 bits of a generic pointer into __shared__ are the hardware LDS
// address (ISA 10.2, LDS aperture: the low 32 address bits are used).
static __device__ __forceinline__ unsigned lds_addr32(const void* p) {
    return (unsigned)(uintptr_t)p;
}

// Async copy 32B (two b128) global -> LDS. INST_OFFSET is added to BOTH the
// LDS and the memory address (ISA 10.7 async pseudocode), so one LDS-address
// VGPR covers both halves. Tracked by ASYNCcnt.
static __device__ __forceinline__ void async_cp32(unsigned lds, const half_t* g) {
    asm volatile("global_load_async_to_lds_b128 %0, %1, off"
                 :: "v"(lds), "v"(g) : "memory");
    asm volatile("global_load_async_to_lds_b128 %0, %1, off offset:16"
                 :: "v"(lds), "v"(g) : "memory");
}

// ---------------- meta / prep kernels ----------------

__global__ void k_zero_meta(int* meta) {
    int t = threadIdx.x;
    if (t <= M_MAXN) meta[t] = 0;
}

__global__ void k_count(const int* __restrict__ idx, int* __restrict__ meta) {
    int i = blockIdx.x * blockDim.x + threadIdx.x;
    if (i < NROWS) atomicAdd(&meta[M_COUNTS + idx[i]], 1);
}

__global__ void k_scan(int* meta) {
    if (threadIdx.x == 0) {
        int acc = 0, mx = 0;
        for (int b = 0; b < B; b++) {
            int c = meta[M_COUNTS + b];
            meta[M_STARTS + b] = acc;
            acc += c;
            if (c > mx) mx = c;
        }
        meta[M_MAXN] = mx;
    }
}

__global__ void k_zero16(uint4* __restrict__ p, size_t n) {
    size_t i = (size_t)blockIdx.x * blockDim.x + threadIdx.x;
    size_t st = (size_t)gridDim.x * blockDim.x;
    uint4 z; z.x = z.y = z.z = z.w = 0u;
    for (; i < n; i += st) p[i] = z;
}

__global__ void k_cvt(const float* __restrict__ s, half_t* __restrict__ d, int n) {
    int i = blockIdx.x * blockDim.x + threadIdx.x;
    int st = gridDim.x * blockDim.x;
    for (; i < n; i += st) d[i] = (half_t)s[i];
}

__global__ __launch_bounds__(128)
void k_scatter(const float* __restrict__ emb, const int* __restrict__ idx,
               const int* __restrict__ meta, float* __restrict__ Xf,
               half_t* __restrict__ Xh) {
    const int i = blockIdx.x;
    const int g = idx[i];
    const int p = i - meta[M_STARTS + g];   // idx is sorted (reference setup)
    const size_t dst = ((size_t)g * CAP + p) * D;
    const int t = threadIdx.x;
    float4 v = *(const float4*)(emb + (size_t)i * D + t * 4);
    *(float4*)(Xf + dst + t * 4) = v;
    union { half_t h[4]; uint2 u; } pk;
    pk.h[0] = (half_t)v.x; pk.h[1] = (half_t)v.y;
    pk.h[2] = (half_t)v.z; pk.h[3] = (half_t)v.w;
    *(uint2*)(Xh + dst + t * 4) = pk.u;
}

// ---------------- WMMA GEMM: OUT[s,m,:] = A[s,m,:] @ W^T (+bias, +relu/resid) ----
// MODE 0: f16 out; MODE 1: relu -> f16 out; MODE 2: +resid -> f32 out
template<int KDIM, int MODE>
__global__ __launch_bounds__(256)
void k_gemm(const half_t* __restrict__ A, const half_t* __restrict__ W,
            const float* __restrict__ bias, const float* __restrict__ resid,
            void* __restrict__ outp, int ldo, const int* __restrict__ meta) {
    __shared__ half_t As[128][48];
    __shared__ half_t Bs[128][48];
    const int maxn = meta[M_MAXN];
    const int m0 = blockIdx.x * 128;
    if (m0 >= maxn) return;
    const int n0 = blockIdx.y * 128;
    const int s  = blockIdx.z;
    const int tid = threadIdx.x;
    const int lane = tid & 31;
    const int wave = tid >> 5;
    const int Mw = (wave & 1) * 64, Nw = (wave >> 1) * 32;
    const int m  = lane & 15, hs = lane >> 4;

    const half_t* Ab = A + ((size_t)s * CAP + m0) * KDIM;
    const half_t* Wb = W + (size_t)n0 * KDIM;

    v8f acc[4][2];
#pragma unroll
    for (int i = 0; i < 4; i++)
#pragma unroll
        for (int j = 0; j < 2; j++) acc[i][j] = (v8f){0,0,0,0,0,0,0,0};

    const int lr = tid >> 1;
    const int lc = (tid & 1) * 16;
    const unsigned la = lds_addr32(&As[lr][lc]);
    const unsigned lb = lds_addr32(&Bs[lr][lc]);

    for (int kk = 0; kk < KDIM; kk += 32) {
        // CDNA5 async global->LDS staging (ASYNCcnt), no VGPR round-trip.
        async_cp32(la, Ab + (size_t)lr * KDIM + kk + lc);
        async_cp32(lb, Wb + (size_t)lr * KDIM + kk + lc);
        asm volatile("s_wait_asynccnt 0" ::: "memory");
        __syncthreads();

        union { v16h v; unsigned u[8]; } aF[4], bF[2];
#pragma unroll
        for (int v = 0; v < 8; v++) {
            const int ka = ((v & 3) * 2) + ((v >> 2) * 16) + hs * 8;  // A-frag k map
            const int kb = ((v & 3) * 2) + ((v >> 2) * 16) + hs * 8;  // B-frag k map
#pragma unroll
            for (int i = 0; i < 4; i++) aF[i].u[v] = ldsu(&As[Mw + i * 16 + m][ka]);
#pragma unroll
            for (int j = 0; j < 2; j++) bF[j].u[v] = ldsu(&Bs[Nw + j * 16 + m][kb]);
        }
#pragma unroll
        for (int i = 0; i < 4; i++)
#pragma unroll
            for (int j = 0; j < 2; j++)
                acc[i][j] = __builtin_amdgcn_wmma_f32_16x16x32_f16(
                    false, aF[i].v, false, bF[j].v, (short)0, acc[i][j], false, false);
        __syncthreads();
    }

#pragma unroll
    for (int i = 0; i < 4; i++) {
#pragma unroll
        for (int j = 0; j < 2; j++) {
            const int col = n0 + Nw + j * 16 + m;
            const float bv = bias[col];
#pragma unroll
            for (int r = 0; r < 8; r++) {
                const size_t rowg = (size_t)s * CAP + m0 + Mw + i * 16 + r + hs * 8;
                float vv = acc[i][j][r] + bv;
                if constexpr (MODE == 1) vv = fmaxf(vv, 0.0f);
                if constexpr (MODE == 2) {
                    vv += resid[rowg * D + col];
                    ((float*)outp)[rowg * (size_t)ldo + col] = vv;
                } else {
                    ((half_t*)outp)[rowg * (size_t)ldo + col] = (half_t)vv;
                }
            }
        }
    }
}

// ---------------- attention: seq axis = 64 studies, per (n, head) ----------------
__global__ __launch_bounds__(256)
void k_attn(const half_t* __restrict__ qkv, const int* __restrict__ meta,
            half_t* __restrict__ AOh) {
    __shared__ half_t P[8][64][72];
    const int n = blockIdx.x;
    const int maxn = meta[M_MAXN];
    if (n >= maxn) return;
    const int tid = threadIdx.x, lane = tid & 31, h = tid >> 5;
    const int m = lane & 15, hs = lane >> 4;
    int cnt[4];
#pragma unroll
    for (int j = 0; j < 4; j++) cnt[j] = meta[M_COUNTS + j * 16 + m];

    v8f S[4][4];
#pragma unroll
    for (int i = 0; i < 4; i++)
#pragma unroll
        for (int j = 0; j < 4; j++) S[i][j] = (v8f){0,0,0,0,0,0,0,0};

    // S = Q @ K^T  (K-dim = e = 64)
#pragma unroll
    for (int ks = 0; ks < 2; ks++) {
        union { v16h v; unsigned u[8]; } qF[4], kF[4];
#pragma unroll
        for (int v = 0; v < 8; v++) {
            const int ea = ((v & 3) * 2) + ((v >> 2) * 16) + hs * 8 + ks * 32;
            const int eb = ((v & 3) * 2) + ((v >> 2) * 16) + hs * 8 + ks * 32;
#pragma unroll
            for (int i = 0; i < 4; i++) {
                const size_t srow = (size_t)(i * 16 + m) * CAP + n;
                qF[i].u[v] = ldsu(qkv + srow * 1536 + h * 64 + ea);
            }
#pragma unroll
            for (int j = 0; j < 4; j++) {
                const size_t trow = (size_t)(j * 16 + m) * CAP + n;
                kF[j].u[v] = ldsu(qkv + trow * 1536 + 512 + h * 64 + eb);
            }
        }
#pragma unroll
        for (int i = 0; i < 4; i++)
#pragma unroll
            for (int j = 0; j < 4; j++)
                S[i][j] = __builtin_amdgcn_wmma_f32_16x16x32_f16(
                    false, qF[i].v, false, kF[j].v, (short)0, S[i][j], false, false);
    }

    // masked softmax over t (cols: 4 tiles x 16 lanes), scale 1/sqrt(64)
    const float scale = 0.125f;
#pragma unroll
    for (int i = 0; i < 4; i++) {
#pragma unroll
        for (int r = 0; r < 8; r++) {
            float sv[4];
            float mx = -__builtin_inff();
#pragma unroll
            for (int j = 0; j < 4; j++) {
                float x = S[i][j][r] * scale;
                if (n >= cnt[j]) x = -__builtin_inff();
                sv[j] = x;
                mx = fmaxf(mx, x);
            }
#pragma unroll
            for (int off = 1; off < 16; off <<= 1)
                mx = fmaxf(mx, __shfl_xor(mx, off, 32));
            float sum = 0.f;
#pragma unroll
            for (int j = 0; j < 4; j++) { sv[j] = __expf(sv[j] - mx); sum += sv[j]; }
#pragma unroll
            for (int off = 1; off < 16; off <<= 1)
                sum += __shfl_xor(sum, off, 32);
            const float inv = 1.0f / sum;
            const int srow = i * 16 + r + hs * 8;
#pragma unroll
            for (int j = 0; j < 4; j++)
                P[h][srow][j * 16 + m] = (half_t)(sv[j] * inv);
        }
    }
    asm volatile("s_wait_dscnt 0" ::: "memory");  // P writes visible to own wave reads

    // AO = P @ V  (K-dim = t = 64)
    v8f O[4][4];
#pragma unroll
    for (int i = 0; i < 4; i++)
#pragma unroll
        for (int j = 0; j < 4; j++) O[i][j] = (v8f){0,0,0,0,0,0,0,0};

#pragma unroll
    for (int ks = 0; ks < 2; ks++) {
        union { v16h v; unsigned u[8]; } pF[4];
        union { v16h v; half_t hh[16]; } vF[4];
#pragma unroll
        for (int v = 0; v < 8; v++) {
            const int ta = ((v & 3) * 2) + ((v >> 2) * 16) + hs * 8 + ks * 32;
            const int tb = ((v & 3) * 2) + ((v >> 2) * 16) + hs * 8 + ks * 32;
#pragma unroll
            for (int i = 0; i < 4; i++)
                pF[i].u[v] = ldsu(&P[h][i * 16 + m][ta]);
#pragma unroll
            for (int j = 0; j < 4; j++) {
                const int e = j * 16 + m;
                vF[j].hh[2 * v]     = qkv[((size_t)tb * CAP + n) * 1536 + 1024 + h * 64 + e];
                vF[j].hh[2 * v + 1] = qkv[((size_t)(tb + 1) * CAP + n) * 1536 + 1024 + h * 64 + e];
            }
        }
#pragma unroll
        for (int i = 0; i < 4; i++)
#pragma unroll
            for (int j = 0; j < 4; j++)
                O[i][j] = __builtin_amdgcn_wmma_f32_16x16x32_f16(
                    false, pF[i].v, false, vF[j].v, (short)0, O[i][j], false, false);
    }

#pragma unroll
    for (int i = 0; i < 4; i++)
#pragma unroll
        for (int j = 0; j < 4; j++)
#pragma unroll
            for (int r = 0; r < 8; r++) {
                const int srow = i * 16 + r + hs * 8;
                const int d = h * 64 + j * 16 + m;
                AOh[((size_t)srow * CAP + n) * D + d] = (half_t)O[i][j][r];
            }
}

// ---------------- LayerNorm over D=512 ----------------
__global__ __launch_bounds__(256)
void k_ln(const float* __restrict__ Z, const float* __restrict__ g,
          const float* __restrict__ bta, float* __restrict__ Xf,
          half_t* __restrict__ Xh, const int* __restrict__ meta) {
    const int n = blockIdx.x;
    if (n >= meta[M_MAXN]) return;
    const int s = blockIdx.y;
    const size_t base = ((size_t)s * CAP + n) * D;
    const int tid = threadIdx.x;
    float x0 = Z[base + tid], x1 = Z[base + tid + 256];
    __shared__ float red[256];
    red[tid] = x0 + x1;
    __syncthreads();
    for (int st = 128; st > 0; st >>= 1) {
        if (tid < st) red[tid] += red[tid + st];
        __syncthreads();
    }
    const float mean = red[0] * (1.0f / D);
    __syncthreads();
    const float d0 = x0 - mean, d1 = x1 - mean;
    red[tid] = d0 * d0 + d1 * d1;
    __syncthreads();
    for (int st = 128; st > 0; st >>= 1) {
        if (tid < st) red[tid] += red[tid + st];
        __syncthreads();
    }
    const float rstd = rsqrtf(red[0] * (1.0f / D) + 1e-5f);
    const float y0 = d0 * rstd * g[tid] + bta[tid];
    const float y1 = d1 * rstd * g[tid + 256] + bta[tid + 256];
    Xf[base + tid] = y0;
    Xf[base + tid + 256] = y1;
    if (Xh) {
        Xh[base + tid] = (half_t)y0;
        Xh[base + tid + 256] = (half_t)y1;
    }
}

// ---------------- mean pool over n (incl. pads), divide by max_n --------------
__global__ __launch_bounds__(512)
void k_mean(const float* __restrict__ Xf, const int* __restrict__ meta,
            float* __restrict__ out) {
    const int s = blockIdx.x, d = threadIdx.x;
    const int maxn = meta[M_MAXN];
    const float* p = Xf + (size_t)s * CAP * D + d;
    float sum = 0.f;
    for (int n = 0; n < maxn; n++) sum += p[(size_t)n * D];
    out[s * D + d] = sum / (float)maxn;
}

// ---------------- driver ----------------
extern "C" void kernel_launch(void* const* d_in, const int* in_sizes, int n_in,
                              void* d_out, int out_size, void* d_ws, size_t ws_size,
                              hipStream_t stream) {
    (void)in_sizes; (void)n_in; (void)out_size; (void)ws_size;
    const float* emb  = (const float*)d_in[0];
    const int*   sidx = (const int*)d_in[1];
    const float* W_in = (const float*)d_in[2];
    const float* b_in = (const float*)d_in[3];
    const float* W_o  = (const float*)d_in[4];
    const float* b_o  = (const float*)d_in[5];
    const float* ln1g = (const float*)d_in[6];
    const float* ln1b = (const float*)d_in[7];
    const float* W1   = (const float*)d_in[8];
    const float* b1   = (const float*)d_in[9];
    const float* W2   = (const float*)d_in[10];
    const float* b2   = (const float*)d_in[11];
    const float* ln2g = (const float*)d_in[12];
    const float* ln2b = (const float*)d_in[13];

    char* ws = (char*)d_ws;
    int*    meta = (int*)(ws + OFF_META);
    half_t* WinH = (half_t*)(ws + OFF_WINH);
    half_t* WoH  = (half_t*)(ws + OFF_WOH);
    half_t* W1H  = (half_t*)(ws + OFF_W1H);
    half_t* W2H  = (half_t*)(ws + OFF_W2H);
    float*  Xf   = (float*)(ws + OFF_XF);
    half_t* Xh   = (half_t*)(ws + OFF_XH);
    half_t* QKVh = (half_t*)(ws + OFF_QKV);
    half_t* AOh  = (half_t*)(ws + OFF_AO);
    half_t* H1h  = (half_t*)(ws + OFF_H1);
    float*  Zf   = (float*)(ws + OFF_ZF);

    k_zero_meta<<<1, 256, 0, stream>>>(meta);
    k_count<<<NROWS / 256, 256, 0, stream>>>(sidx, meta);
    k_scan<<<1, 64, 0, stream>>>(meta);

    k_cvt<<<1024, 256, 0, stream>>>(W_in, WinH, 3 * D * D);
    k_cvt<<<512, 256, 0, stream>>>(W_o, WoH, D * D);
    k_cvt<<<1024, 256, 0, stream>>>(W1, W1H, 2 * D * D);
    k_cvt<<<1024, 256, 0, stream>>>(W2, W2H, 2 * D * D);

    k_zero16<<<2048, 256, 0, stream>>>((uint4*)Xf, SZ_XF / 16);
    k_zero16<<<2048, 256, 0, stream>>>((uint4*)Xh, SZ_XH / 16);
    k_scatter<<<NROWS, 128, 0, stream>>>(emb, sidx, meta, Xf, Xh);

    dim3 blk(256);
    dim3 g_qkv(CAP / 128, (3 * D) / 128, B);
    k_gemm<512, 0><<<g_qkv, blk, 0, stream>>>(Xh, WinH, b_in, nullptr, QKVh, 3 * D, meta);

    k_attn<<<CAP, 256, 0, stream>>>(QKVh, meta, AOh);

    dim3 g_d(CAP / 128, D / 128, B);
    k_gemm<512, 2><<<g_d, blk, 0, stream>>>(AOh, WoH, b_o, Xf, Zf, D, meta);
    k_ln<<<dim3(CAP, B), 256, 0, stream>>>(Zf, ln1g, ln1b, Xf, Xh, meta);

    dim3 g_ff(CAP / 128, (2 * D) / 128, B);
    k_gemm<512, 1><<<g_ff, blk, 0, stream>>>(Xh, W1H, b1, nullptr, H1h, 2 * D, meta);
    k_gemm<1024, 2><<<g_d, blk, 0, stream>>>(H1h, W2H, b2, Xf, Zf, D, meta);
    k_ln<<<dim3(CAP, B), 256, 0, stream>>>(Zf, ln2g, ln2b, Xf, nullptr, meta);

    k_mean<<<B, 512, 0, stream>>>(Xf, meta, (float*)d_out);
}
